// GroupedMoE_25615184953558
// MI455X (gfx1250) — compile-verified, matrix operand order checked by reading, not run on hardware
//
#include <hip/hip_runtime.h>
#include <stdint.h>

// ---------------------------------------------------------------------------
// GroupedMoE (top-2 of 8 experts, swiglu) for MI455X / gfx1250, wave32 + WMMA.
//
// Pipeline per launch:
//   1) memset y = 0
//   2) gate kernel  -> coeff[T, E]  (softmax probs scattered at top-2 slots)
//   3) bf16 pre-swizzle of x, w1g, w1u, w2 into per-lane WMMA fragment order
//   4) fused FFN kernel: per (16-token tile, expert):
//        x tile staged to LDS with TENSOR_LOAD_TO_LDS (TDM) + s_wait_tensorcnt
//        g,u = x @ w1{g,u}[e]   (v_wmma_f32_16x16x32_bf16)
//        h'  = coeff * silu(g) * u  -> bf16 LDS (padded rows, v_rcp silu)
//        y  += h' @ w2[e]           (WMMA, fp32 global atomics combine experts)
//
// Workspace layout (bytes):    needs ~74 MB
//   [0          , 65536)       coeff fp32 [2048][8]
//   [65536      , +4194304)    xb   bf16 fragments  [T/16][D/32][1KB]
//   [..         , +23068672)   w1gb bf16 fragments  [E][H/16][D/32][1KB]
//   [..         , +23068672)   w1ub
//   [..         , +23068672)   w2b  bf16 fragments  [E][D/16][H/32][1KB]
// ---------------------------------------------------------------------------

#define T_TOK 2048
#define DDIM  1024
#define HDIM  1408
#define NEXP  8
#define HT_TILES 88   // H/16
#define KB_D     32   // D/32
#define DT_TILES 64   // D/16
#define HB_TILES 44   // H/32
#define RS     1416   // padded h' row stride in bf16 elems (1408 + 8)

typedef __attribute__((ext_vector_type(16))) __bf16 v16bf;
typedef __attribute__((ext_vector_type(8)))  float  v8f;
typedef __attribute__((ext_vector_type(4)))  unsigned int u32x4;
typedef __attribute__((ext_vector_type(8)))  int  i32x8;
typedef __attribute__((ext_vector_type(4)))  int  i32x4;

union Frag {
  v16bf v;
  uint4 u[2];
};

__device__ inline void atomAddF(float* p, float v) {
#if __has_builtin(__builtin_amdgcn_global_atomic_fadd_f32)
  __builtin_amdgcn_global_atomic_fadd_f32(p, v);
#else
  atomicAdd(p, v);
#endif
}

// ---------------------------------------------------------------------------
// TDM: 1-D contiguous copy global -> LDS, `bytes` must be multiple of 4.
// Builds a raw D# (ISA 08_async_tensor §8.3/8.4): count=1, type=2("image"),
// data_size=4B, single row of tile_dim0 = bytes/4 elements.
// ---------------------------------------------------------------------------
__device__ inline void tdm_copy_to_lds(const void* gsrc, void* lds_dst,
                                       unsigned int bytes) {
  const unsigned long long ga = (unsigned long long)(size_t)gsrc;
  const unsigned int lds_addr = (unsigned int)(size_t)lds_dst;  // addr[31:0]
  const unsigned int elems = bytes >> 2;

  u32x4 g0;
  g0[0] = 1u;                                    // count=1, no gather
  g0[1] = lds_addr;                              // lds_addr [63:32]
  g0[2] = (unsigned int)ga;                      // global_addr lo
  g0[3] = (unsigned int)(ga >> 32) | (2u << 30); // global_addr hi | type=2

  i32x8 g1;
  g1[0] = (int)(2u << 16);                 // wg_mask=0 | data_size=2 (4B)
  g1[1] = (int)((elems & 0xFFFFu) << 16);  // tensor_dim0 lo16 at [63:48]
  g1[2] = (int)(((elems >> 16) & 0xFFFFu) | (1u << 16)); // dim0 hi | tensor_dim1=1
  g1[3] = (int)((elems & 0xFFFFu) << 16);  // tile_dim0 at [127:112]
  g1[4] = 1;                               // tile_dim1 = 1
  g1[5] = (int)elems;                      // tensor_dim0_stride lo32
  g1[6] = 0;
  g1[7] = 0;

  i32x4 z4 = {0, 0, 0, 0};
#if __clang_major__ >= 23
  i32x8 z8 = {0, 0, 0, 0, 0, 0, 0, 0};
  __builtin_amdgcn_tensor_load_to_lds(g0, g1, z4, z4, z8, 0);
#else
  __builtin_amdgcn_tensor_load_to_lds(g0, g1, z4, z4, 0);
#endif
}

// ---------------------------------------------------------------------------
// Gate: logits = x @ gate_w^T, softmax over 8, top-2 -> coeff[T,8]
// ---------------------------------------------------------------------------
__global__ __launch_bounds__(256) void moe_gate_kernel(
    const float* __restrict__ x, const float* __restrict__ gate_w,
    float* __restrict__ coeff) {
  __shared__ float sgw[NEXP * DDIM];  // 32 KB
  const int tid = threadIdx.x;
  for (int i = tid; i < NEXP * DDIM; i += 256) sgw[i] = gate_w[i];
  __syncthreads();

  const int t = blockIdx.x * 256 + tid;
  if (t >= T_TOK) return;
  const float4* xt = (const float4*)(x + (size_t)t * DDIM);

  float acc[NEXP];
#pragma unroll
  for (int e = 0; e < NEXP; ++e) acc[e] = 0.f;
  for (int d4 = 0; d4 < DDIM / 4; ++d4) {
    float4 xv = xt[d4];
    const int d = d4 * 4;
#pragma unroll
    for (int e = 0; e < NEXP; ++e) {
      const float* w = &sgw[e * DDIM + d];
      acc[e] += xv.x * w[0] + xv.y * w[1] + xv.z * w[2] + xv.w * w[3];
    }
  }
  float m = acc[0];
#pragma unroll
  for (int e = 1; e < NEXP; ++e) m = fmaxf(m, acc[e]);
  float s[NEXP], sum = 0.f;
#pragma unroll
  for (int e = 0; e < NEXP; ++e) { s[e] = __expf(acc[e] - m); sum += s[e]; }
  const float inv = __builtin_amdgcn_rcpf(sum);
#pragma unroll
  for (int e = 0; e < NEXP; ++e) s[e] *= inv;

  int i1 = 0;
#pragma unroll
  for (int e = 1; e < NEXP; ++e) if (s[e] > s[i1]) i1 = e;
  int i2 = (i1 == 0) ? 1 : 0;
#pragma unroll
  for (int e = 0; e < NEXP; ++e) if (e != i1 && s[e] > s[i2]) i2 = e;
#pragma unroll
  for (int e = 0; e < NEXP; ++e)
    coeff[(size_t)t * NEXP + e] = (e == i1) ? s[i1] : (e == i2) ? s[i2] : 0.f;
}

// ---------------------------------------------------------------------------
// Fragment layout (per 16x32 bf16 tile, 1KB blob, lane-major):
//   lane l: row = l&15, half = l>>4
//   bf16[0..7]  = tile[row][half*8 + 0..7]
//   bf16[8..15] = tile[row][16 + half*8 + 0..7]
// A tiles: row = M (token).  B tiles: row = N (output col), k-major source.
// ---------------------------------------------------------------------------

__global__ __launch_bounds__(128) void cvt_x_kernel(
    const float* __restrict__ x, char* __restrict__ xb) {
  const int f = blockIdx.x * 4 + (threadIdx.x >> 5);  // 4096 fragments
  const int lane = threadIdx.x & 31;
  const int t16 = f >> 5, kb = f & 31;
  const int row = lane & 15, half = lane >> 4;
  const float* src = x + (size_t)(t16 * 16 + row) * DDIM + kb * 32 + half * 8;
  Frag fr;
#pragma unroll
  for (int i = 0; i < 8; ++i) {
    fr.v[i]     = (__bf16)src[i];
    fr.v[8 + i] = (__bf16)src[16 + i];
  }
  char* dst = xb + (size_t)f * 1024 + lane * 32;
  *(uint4*)dst        = fr.u[0];
  *(uint4*)(dst + 16) = fr.u[1];
}

// w1 source: [E][D][H]; B tile (ht,kb): N = h, K = d
__global__ __launch_bounds__(128) void cvt_w1_kernel(
    const float* __restrict__ w, char* __restrict__ dst) {
  const int f = blockIdx.x * 4 + (threadIdx.x >> 5);  // 22528 fragments
  const int lane = threadIdx.x & 31;
  const int e = f / 2816, r = f % 2816;
  const int ht = r >> 5, kb = r & 31;
  const int n = lane & 15, half = lane >> 4;
  const int h = ht * 16 + n;
  const int d0 = kb * 32 + half * 8;
  const float* base = w + (size_t)e * DDIM * HDIM + h;
  Frag fr;
#pragma unroll
  for (int i = 0; i < 8; ++i) {
    fr.v[i]     = (__bf16)base[(size_t)(d0 + i) * HDIM];
    fr.v[8 + i] = (__bf16)base[(size_t)(d0 + 16 + i) * HDIM];
  }
  char* p = dst + (size_t)f * 1024 + lane * 32;
  *(uint4*)p        = fr.u[0];
  *(uint4*)(p + 16) = fr.u[1];
}

// w2 source: [E][H][D]; B tile (dt,hb): N = d, K = h
__global__ __launch_bounds__(128) void cvt_w2_kernel(
    const float* __restrict__ w, char* __restrict__ dst) {
  const int f = blockIdx.x * 4 + (threadIdx.x >> 5);  // 22528 fragments
  const int lane = threadIdx.x & 31;
  const int e = f / 2816, r = f % 2816;
  const int dt = r / 44, hb = r % 44;
  const int n = lane & 15, half = lane >> 4;
  const int d = dt * 16 + n;
  const int h0 = hb * 32 + half * 8;
  const float* base = w + (size_t)e * HDIM * DDIM + d;
  Frag fr;
#pragma unroll
  for (int i = 0; i < 8; ++i) {
    fr.v[i]     = (__bf16)base[(size_t)(h0 + i) * DDIM];
    fr.v[8 + i] = (__bf16)base[(size_t)(h0 + 16 + i) * DDIM];
  }
  char* p = dst + (size_t)f * 1024 + lane * 32;
  *(uint4*)p        = fr.u[0];
  *(uint4*)(p + 16) = fr.u[1];
}

// ---------------------------------------------------------------------------
// Fused FFN: one block = (16-token tile, expert), 4 waves.
// Dynamic LDS: [0,32768) x-tile blob | [32768,+45312) h' bf16 | 16 coeffs
// ---------------------------------------------------------------------------
__global__ __launch_bounds__(128) void moe_ffn_kernel(
    const char* __restrict__ xb, const char* __restrict__ w1gb,
    const char* __restrict__ w1ub, const char* __restrict__ w2b,
    const float* __restrict__ coeff, float* __restrict__ y) {
  extern __shared__ char smem[];
  char*   s_x = smem;                                // 32768 B
  __bf16* s_h = (__bf16*)(smem + 32768);             // 45312 B
  float*  s_c = (float*)(smem + 32768 + 45312);      // 64 B

  const int tt   = blockIdx.x;
  const int e    = blockIdx.y;
  const int tid  = threadIdx.x;
  const int wave = tid >> 5;
  const int lane = tid & 31;
  const int n    = lane & 15, half = lane >> 4;

  if (tid < 16) s_c[tid] = coeff[(size_t)(tt * 16 + tid) * NEXP + e];
  __syncthreads();
  bool any = false;
#pragma unroll
  for (int i = 0; i < 16; ++i) any |= (s_c[i] != 0.f);
  if (!any) return;  // uniform across block: tile uses this expert nowhere

  // Stage x-tile fragments (32 KB) into LDS via the Tensor Data Mover.
  if (wave == 0) {
    tdm_copy_to_lds(xb + (size_t)tt * 32 * 1024, s_x, 32 * 1024);
    __builtin_amdgcn_s_wait_tensorcnt(0);
  }
  __syncthreads();

  // ---- Stage 1: g,u = x @ w1{g,u}[e]; h' = coeff*silu(g)*u -> LDS ----
  const char* wgBase = w1gb + (size_t)e * 2816 * 1024 + (size_t)lane * 32;
  const char* wuBase = w1ub + (size_t)e * 2816 * 1024 + (size_t)lane * 32;
  for (int it = 0; it < HT_TILES / 4; ++it) {
    const int ht = it * 4 + wave;
    v8f accg = {};
    v8f accu = {};
    const char* wg = wgBase + (size_t)ht * 32 * 1024;
    const char* wu = wuBase + (size_t)ht * 32 * 1024;
    const char* ax = s_x + lane * 32;
    // prefetch next h-tile's weight streams (4 tiles = 128KB ahead)
    __builtin_prefetch(wg + 4 * 32 * 1024, 0, 1);
    __builtin_prefetch(wu + 4 * 32 * 1024, 0, 1);
#pragma unroll 4
    for (int kb = 0; kb < KB_D; ++kb) {
      Frag a, bg, bu;
      a.u[0]  = *(const uint4*)(ax);
      a.u[1]  = *(const uint4*)(ax + 16);
      bg.u[0] = *(const uint4*)(wg);
      bg.u[1] = *(const uint4*)(wg + 16);
      bu.u[0] = *(const uint4*)(wu);
      bu.u[1] = *(const uint4*)(wu + 16);
      accg = __builtin_amdgcn_wmma_f32_16x16x32_bf16(false, a.v, false, bg.v,
                                                     (short)0, accg, false, false);
      accu = __builtin_amdgcn_wmma_f32_16x16x32_bf16(false, a.v, false, bu.v,
                                                     (short)0, accu, false, false);
      ax += 1024; wg += 1024; wu += 1024;
    }
    // C layout: VGPR j holds (row = j + 8*half, col = n)
#pragma unroll
    for (int j = 0; j < 8; ++j) {
      const int row = j + 8 * half;
      const float c = s_c[row];
      const float g = accg[j];
      const float u = accu[j];
      // coeff * silu(g) * u, fast sigmoid via v_rcp_f32
      const float sig = __builtin_amdgcn_rcpf(1.f + __expf(-g));
      const float hh = c * g * sig * u;
      s_h[row * RS + ht * 16 + n] = (__bf16)hh;
    }
  }
  __syncthreads();

  // ---- Stage 2: y[tile] += h' @ w2[e] ----
  const char* w2Base = w2b + (size_t)e * 2816 * 1024 + (size_t)lane * 32;
  for (int it = 0; it < DT_TILES / 4; ++it) {
    const int dt = it * 4 + wave;
    v8f acc = {};
    const char* wb = w2Base + (size_t)dt * 44 * 1024;
    const __bf16* ah = s_h + n * RS + half * 8;
    __builtin_prefetch(wb + 4 * 44 * 1024, 0, 1);
#pragma unroll 4
    for (int hb = 0; hb < HB_TILES; ++hb) {
      Frag a, b;
      a.u[0] = *(const uint4*)(ah + hb * 32);
      a.u[1] = *(const uint4*)(ah + hb * 32 + 16);
      b.u[0] = *(const uint4*)(wb);
      b.u[1] = *(const uint4*)(wb + 16);
      acc = __builtin_amdgcn_wmma_f32_16x16x32_bf16(false, a.v, false, b.v,
                                                    (short)0, acc, false, false);
      wb += 1024;
    }
#pragma unroll
    for (int j = 0; j < 8; ++j) {
      const int row = tt * 16 + j + 8 * half;
      const int col = dt * 16 + n;
      atomAddF(&y[(size_t)row * DDIM + col], acc[j]);
    }
  }
}

// ---------------------------------------------------------------------------
extern "C" void kernel_launch(void* const* d_in, const int* in_sizes, int n_in,
                              void* d_out, int out_size, void* d_ws, size_t ws_size,
                              hipStream_t stream) {
  (void)in_sizes; (void)n_in; (void)ws_size;
  const float* x      = (const float*)d_in[0];
  const float* gate_w = (const float*)d_in[1];
  const float* w1g    = (const float*)d_in[2];
  const float* w1u    = (const float*)d_in[3];
  const float* w2     = (const float*)d_in[4];
  float* y = (float*)d_out;

  char* ws    = (char*)d_ws;
  float* coeff = (float*)ws;
  char* xb    = ws + 65536;
  char* w1gb  = xb + 4194304;
  char* w1ub  = w1gb + 23068672;
  char* w2b   = w1ub + 23068672;

  hipMemsetAsync(d_out, 0, (size_t)out_size * sizeof(float), stream);

  moe_gate_kernel<<<T_TOK / 256, 256, 0, stream>>>(x, gate_w, coeff);
  cvt_x_kernel<<<1024, 128, 0, stream>>>(x, xb);
  cvt_w1_kernel<<<5632, 128, 0, stream>>>(w1g, w1gb);
  cvt_w1_kernel<<<5632, 128, 0, stream>>>(w1u, w1ub);
  cvt_w2_kernel<<<5632, 128, 0, stream>>>(w2, w2b);

  dim3 grid(T_TOK / 16, NEXP);
  const size_t shmem = 32768 + 45312 + 64;
  moe_ffn_kernel<<<grid, 128, shmem, stream>>>(xb, w1gb, w1ub, w2b, coeff, y);
}
